// LSTM_74672301408953
// MI455X (gfx1250) — matrix operand order repaired and checked by reading, not compiled, not run on hardware
//
#include <hip/hip_runtime.h>
#include <hip/hip_bf16.h>

// ---------------------------------------------------------------------------
// Feature probes (keep compile safe across toolchains)
// ---------------------------------------------------------------------------
#if defined(__has_builtin)
#  if __has_builtin(__builtin_amdgcn_global_load_async_to_lds_b128)
#    define HAVE_ASYNC_LDS 1
#  else
#    define HAVE_ASYNC_LDS 0
#  endif
#  if __has_builtin(__builtin_amdgcn_s_wait_asynccnt)
#    define HAVE_WAIT_ASYNC_BUILTIN 1
#  else
#    define HAVE_WAIT_ASYNC_BUILTIN 0
#  endif
#  if __has_builtin(__builtin_amdgcn_tanhf)
#    define HAVE_TANH 1
#  else
#    define HAVE_TANH 0
#  endif
#else
#  define HAVE_ASYNC_LDS 0
#  define HAVE_WAIT_ASYNC_BUILTIN 0
#  define HAVE_TANH 0
#endif

// ---------------------------------------------------------------------------
// Types for CDNA5 WMMA (wave32, v_wmma_f32_16x16x32_bf16)
// ---------------------------------------------------------------------------
typedef __bf16 bf16_t;
typedef __attribute__((ext_vector_type(16))) __bf16 v16bf;   // 8 VGPRs
typedef __attribute__((ext_vector_type(8)))  float  v8f;     // 8 VGPRs
typedef __attribute__((ext_vector_type(4)))  unsigned int u32x4;

// Pointer types matching __builtin_amdgcn_global_load_async_to_lds_b128:
// (int4 AS1*, int4 AS3*, int offset, int cpol)
typedef __attribute__((__vector_size__(4 * sizeof(int)))) int i32x4_v;
typedef __attribute__((address_space(1))) i32x4_v* gptr_b128;
typedef __attribute__((address_space(3))) i32x4_v* lptr_b128;

union FragBF {
    v16bf v;
    u32x4 q[2];
};

// float -> bf16, round-to-nearest-even
__device__ __forceinline__ bf16_t f2bf(float f) {
    union { float f; unsigned u; } a;
    a.f = f;
    unsigned r = a.u + 0x7FFFu + ((a.u >> 16) & 1u);
    unsigned short h = (unsigned short)(r >> 16);
    return __builtin_bit_cast(bf16_t, h);
}

__device__ __forceinline__ float sigmoidf_fast(float x) {
    // v_exp_f32 + v_rcp_f32, no IEEE-div fixup chain
    return __builtin_amdgcn_rcpf(1.0f + __expf(-x));
}

__device__ __forceinline__ float tanhf_fast(float x) {
#if HAVE_TANH
    return __builtin_amdgcn_tanhf(x);   // CDNA5 v_tanh_f32
#else
    const float xc = fminf(fmaxf(x, -15.0f), 15.0f);
    const float e  = __expf(-2.0f * xc);
    return (1.0f - e) * __builtin_amdgcn_rcpf(1.0f + e);
#endif
}

// A-matrix 16x32 bf16 fragment (ISA 7.12.2): lane m = lane&15, half = lane>>4.
// VGPR0-3: K = half*8 .. half*8+7 ; VGPR4-7: K = 16+half*8 .. 16+half*8+7.
__device__ __forceinline__ v16bf load_frag_A(const bf16_t* base, int ld) {
    const int lane = threadIdx.x & 31;
    const bf16_t* p = base + (lane & 15) * ld + ((lane >> 4) << 3);
    FragBF f;
    f.q[0] = *reinterpret_cast<const u32x4*>(p);
    f.q[1] = *reinterpret_cast<const u32x4*>(p + 16);
    return f.v;
}

// B-matrix 32x16 bf16 fragment: column n = lane&15 (row n of [N,K] weights),
// K = half*8 .. half*8+7, then 16+half*8 .. 16+half*8+7 (matches A).
__device__ __forceinline__ v16bf load_frag_B(const bf16_t* base, int ld) {
    const int lane = threadIdx.x & 31;
    const bf16_t* p = base + (lane & 15) * ld + ((lane >> 4) << 3);
    FragBF f;
    f.q[0] = *reinterpret_cast<const u32x4*>(p);
    f.q[1] = *reinterpret_cast<const u32x4*>(p + 16);
    return f.v;
}

// ---------------------------------------------------------------------------
// Problem constants
// ---------------------------------------------------------------------------
#define T_STEPS 256
#define BATCH   64
#define MDIM    512
#define KDIM    1024            // [X | S]
#define NCAT    2560            // 1536 iou | 512 f | 512 fs
#define ROWS    (T_STEPS * BATCH)   // 16384
#define HPITCH  520             // padded LDS row pitch (bf16): 16 rows -> 16 banks

// Workspace layout (bytes)
#define OFF_A    ((size_t)0)            // 16384*1024 bf16 = 33554432
#define OFF_W    ((size_t)33554432)     // 2560*1024 bf16  = 5242880
#define OFF_WH   ((size_t)38797312)     // 2048*512  bf16  = 2097152
#define OFF_BIAS ((size_t)40894464)     // 2560 f32        = 10240
#define OFF_C    ((size_t)40904704)     // 64*512 f32      = 131072
#define OFF_H0   ((size_t)41035776)     // 64*512 bf16     = 65536
#define OFF_H1   ((size_t)41101312)     // 64*512 bf16     = 65536
#define OFF_PRE  ((size_t)41166848)     // 16384*2560 f32  = 167772160

// ---------------------------------------------------------------------------
// Packing / init kernels
// ---------------------------------------------------------------------------
__global__ __launch_bounds__(256) void pack_A_kernel(
    const float* __restrict__ inputs, const float* __restrict__ sememe_h,
    bf16_t* __restrict__ A)
{
    const int idx = blockIdx.x * 256 + threadIdx.x;   // over ROWS*KDIM
    const int row = idx >> 10;
    const int k   = idx & 1023;
    float v = (k < MDIM) ? inputs[row * MDIM + k]
                         : sememe_h[row * MDIM + (k - MDIM)];
    A[idx] = f2bf(v);
}

__global__ __launch_bounds__(256) void pack_W_kernel(
    const float* __restrict__ W_ioux, const float* __restrict__ W_ious,
    const float* __restrict__ W_fx,   const float* __restrict__ W_fxs,
    const float* __restrict__ W_fs,   bf16_t* __restrict__ W)
{
    const int idx = blockIdx.x * 256 + threadIdx.x;   // over NCAT*KDIM
    const int row = idx >> 10;
    const int k   = idx & 1023;
    float v;
    if (row < 1536) {
        v = (k < MDIM) ? W_ioux[row * MDIM + k] : W_ious[row * MDIM + (k - MDIM)];
    } else if (row < 2048) {
        const int r = row - 1536;
        v = (k < MDIM) ? W_fx[r * MDIM + k] : 0.0f;
    } else {
        const int r = row - 2048;
        v = (k < MDIM) ? W_fxs[r * MDIM + k] : W_fs[r * MDIM + (k - MDIM)];
    }
    W[idx] = f2bf(v);
}

__global__ __launch_bounds__(256) void pack_Wh_kernel(
    const float* __restrict__ W_iouh, const float* __restrict__ W_fh,
    bf16_t* __restrict__ Wh)
{
    const int idx = blockIdx.x * 256 + threadIdx.x;   // over 2048*512
    const int row = idx >> 9;
    const int k   = idx & 511;
    float v = (row < 1536) ? W_iouh[row * MDIM + k] : W_fh[(row - 1536) * MDIM + k];
    Wh[idx] = f2bf(v);
}

__global__ __launch_bounds__(256) void pack_misc_kernel(
    const float* __restrict__ b_ioux, const float* __restrict__ b_ious,
    const float* __restrict__ b_iouh, const float* __restrict__ b_fx,
    const float* __restrict__ b_fh,   const float* __restrict__ b_fxs,
    const float* __restrict__ b_fs,
    float* __restrict__ bias, float* __restrict__ c_state,
    bf16_t* __restrict__ h0, bf16_t* __restrict__ h1)
{
    const int tid = blockIdx.x * 256 + threadIdx.x;   // over 32768
    if (tid < NCAT) {
        float b;
        if (tid < 1536)      b = b_ioux[tid] + b_ious[tid] + b_iouh[tid];
        else if (tid < 2048) b = b_fx[tid - 1536] + b_fh[tid - 1536];
        else                 b = b_fxs[tid - 2048] + b_fs[tid - 2048];
        bias[tid] = b;
    }
    c_state[tid] = 0.0f;
    const bf16_t z = f2bf(0.0f);
    h0[tid] = z;
    h1[tid] = z;
}

// ---------------------------------------------------------------------------
// GEMM 1: Pre[16384,2560] = [X|S] @ Wcat^T + bias; epilogue fuses
// sigmoid(fs)*sememe_c for cols >= 2048.
// Block: 256 threads = 8 waves, block tile 128(M) x 256(N), wave tile 64x64.
// ---------------------------------------------------------------------------
__global__ __launch_bounds__(256) void gemm_pre_kernel(
    const bf16_t* __restrict__ A, const bf16_t* __restrict__ W,
    const float* __restrict__ bias, const float* __restrict__ sememe_c,
    float* __restrict__ Pre)
{
    const int w    = threadIdx.x >> 5;
    const int lane = threadIdx.x & 31;
    const int rowBase = blockIdx.y * 128 + (w >> 2) * 64;
    const int colBase = blockIdx.x * 256 + (w & 3) * 64;

    v8f acc[4][4];
    const v8f vzero = {0.f, 0.f, 0.f, 0.f, 0.f, 0.f, 0.f, 0.f};
    #pragma unroll
    for (int mi = 0; mi < 4; ++mi)
        #pragma unroll
        for (int ni = 0; ni < 4; ++ni)
            acc[mi][ni] = vzero;

    for (int k0 = 0; k0 < KDIM; k0 += 32) {
        __builtin_prefetch(A + (size_t)rowBase * KDIM + k0 + 256, 0, 1);
        v16bf bfr[4];
        #pragma unroll
        for (int ni = 0; ni < 4; ++ni)
            bfr[ni] = load_frag_B(W + (size_t)(colBase + ni * 16) * KDIM + k0, KDIM);
        #pragma unroll
        for (int mi = 0; mi < 4; ++mi) {
            const v16bf afr = load_frag_A(A + (size_t)(rowBase + mi * 16) * KDIM + k0, KDIM);
            #pragma unroll
            for (int ni = 0; ni < 4; ++ni)
                acc[mi][ni] = __builtin_amdgcn_wmma_f32_16x16x32_bf16(
                    false, afr, false, bfr[ni], (short)0, acc[mi][ni], false, false);
        }
    }

    const int half = lane >> 4;
    const int ln   = lane & 15;
    #pragma unroll
    for (int mi = 0; mi < 4; ++mi) {
        #pragma unroll
        for (int ni = 0; ni < 4; ++ni) {
            const int col  = colBase + ni * 16 + ln;
            const float bc = bias[col];
            #pragma unroll
            for (int r = 0; r < 8; ++r) {
                const int row = rowBase + mi * 16 + half * 8 + r;
                float v = acc[mi][ni][r] + bc;
                if (col >= 2048) {
                    v = sigmoidf_fast(v) * sememe_c[(size_t)row * MDIM + (col - 2048)];
                }
                Pre[(size_t)row * NCAT + col] = v;
            }
        }
    }
}

// ---------------------------------------------------------------------------
// Scan step: one launch per t; 8 workgroups, each owns 64 output dims.
// h_in is staged into LDS (async-to-LDS path when available) with a padded
// pitch so A-fragment ds_load_b128 hits 16 distinct banks; the staging
// buffer is reused as the gate buffer after the GEMM.
// ---------------------------------------------------------------------------
__global__ __launch_bounds__(256) void lstm_step_kernel(
    int t, const float* __restrict__ Pre, const bf16_t* __restrict__ Wh,
    const bf16_t* __restrict__ h_in, bf16_t* __restrict__ h_out,
    float* __restrict__ c_state, float* __restrict__ hs_out)
{
    __shared__ __align__(16) unsigned char smem[BATCH * HPITCH * 2]; // 66560 B
    bf16_t* hsh  = (bf16_t*)smem;          // [64][HPITCH] during GEMM
    float*  gbuf = (float*)smem;           // [64][256] after GEMM

    const int j    = blockIdx.x;           // m slice: [j*64, j*64+64)
    const int w    = threadIdx.x >> 5;
    const int lane = threadIdx.x & 31;
    const int tid  = threadIdx.x;
    const int nbase = w * 32;              // wave's 32 local cols

    // ---- Stage h_in (64x512 bf16) into LDS with pitch HPITCH --------------
#if HAVE_ASYNC_LDS
    #pragma unroll
    for (int it = 0; it < 16; ++it) {
        const int e = tid + it * 256;      // 16-byte chunk index, 4096 total
        const int r = e >> 6;              // row 0..63
        const int c = e & 63;              // 16B chunk within row
        __builtin_amdgcn_global_load_async_to_lds_b128(
            (gptr_b128)(void*)(h_in + r * MDIM + c * 8),
            (lptr_b128)(hsh + r * HPITCH + c * 8),
            0, 0);
    }
#if HAVE_WAIT_ASYNC_BUILTIN
    __builtin_amdgcn_s_wait_asynccnt(0);
#else
    asm volatile("s_wait_asynccnt 0" ::: "memory");
#endif
#else
    {
        const u32x4* src = (const u32x4*)h_in;
        u32x4* dst = (u32x4*)smem;
        #pragma unroll
        for (int it = 0; it < 16; ++it) {
            const int e = tid + it * 256;
            const int r = e >> 6;
            const int c = e & 63;
            dst[r * (HPITCH / 8) + c] = src[r * (MDIM / 8) + c];
        }
    }
#endif
    __syncthreads();

    // ---- GEMM: gates[64, 256] = h @ Wh_slice^T ----------------------------
    v8f acc[4][2];
    const v8f vzero = {0.f, 0.f, 0.f, 0.f, 0.f, 0.f, 0.f, 0.f};
    #pragma unroll
    for (int mi = 0; mi < 4; ++mi) {
        acc[mi][0] = vzero;
        acc[mi][1] = vzero;
    }

    int wrowBase[2];
    #pragma unroll
    for (int ni = 0; ni < 2; ++ni) {
        const int cc = nbase + ni * 16;
        const int g  = cc >> 6;                              // gate group 0..3
        wrowBase[ni] = ((g < 3) ? g * MDIM : 1536) + j * 64 + (cc & 63);
    }

    for (int k0 = 0; k0 < MDIM; k0 += 32) {
        v16bf bfr[2];
        #pragma unroll
        for (int ni = 0; ni < 2; ++ni)
            bfr[ni] = load_frag_B(Wh + (size_t)wrowBase[ni] * MDIM + k0, MDIM);
        #pragma unroll
        for (int mi = 0; mi < 4; ++mi) {
            const v16bf afr = load_frag_A(hsh + (mi * 16) * HPITCH + k0, HPITCH);
            #pragma unroll
            for (int ni = 0; ni < 2; ++ni)
                acc[mi][ni] = __builtin_amdgcn_wmma_f32_16x16x32_bf16(
                    false, afr, false, bfr[ni], (short)0, acc[mi][ni], false, false);
        }
    }
    __syncthreads();   // all reads of hsh done; smem becomes gbuf

    const int half = lane >> 4;
    const int ln   = lane & 15;
    #pragma unroll
    for (int mi = 0; mi < 4; ++mi) {
        #pragma unroll
        for (int ni = 0; ni < 2; ++ni) {
            const int col = nbase + ni * 16 + ln;
            #pragma unroll
            for (int r = 0; r < 8; ++r) {
                const int row = mi * 16 + half * 8 + r;   // = batch index b
                gbuf[row * 256 + col] = acc[mi][ni][r];
            }
        }
    }
    __syncthreads();

    // ---- Fused gates + state update ---------------------------------------
    const int tb = t * BATCH;
    #pragma unroll
    for (int it = 0; it < 16; ++it) {
        const int idx = tid + it * 256;   // 0..4095
        const int b   = idx >> 6;
        const int ml  = idx & 63;
        const int m   = j * 64 + ml;
        const float* pr = Pre + (size_t)(tb + b) * NCAT;

        const float xi  = pr[m]          + gbuf[b * 256 + ml];
        const float xo  = pr[512 + m]    + gbuf[b * 256 + 64 + ml];
        const float xu  = pr[1024 + m]   + gbuf[b * 256 + 128 + ml];
        const float xf  = pr[1536 + m]   + gbuf[b * 256 + 192 + ml];
        const float fsc = pr[2048 + m];  // sigmoid(fs_pre) * sememe_c, prefused

        const float ig = sigmoidf_fast(xi);
        const float og = sigmoidf_fast(xo);
        const float ug = tanhf_fast(xu);
        const float fg = sigmoidf_fast(xf);

        const int   cidx = b * MDIM + m;
        const float c    = ig * ug + fg * c_state[cidx] + fsc;
        c_state[cidx] = c;
        const float h = og * tanhf_fast(c);

        hs_out[(size_t)(tb + b) * MDIM + m] = h;
        h_out[cidx] = f2bf(h);
    }
}

// ---------------------------------------------------------------------------
// Host launcher
// ---------------------------------------------------------------------------
extern "C" void kernel_launch(void* const* d_in, const int* in_sizes, int n_in,
                              void* d_out, int out_size, void* d_ws, size_t ws_size,
                              hipStream_t stream) {
    (void)in_sizes; (void)n_in; (void)out_size; (void)ws_size;

    const float* inputs   = (const float*)d_in[0];
    const float* sememe_c = (const float*)d_in[1];
    const float* sememe_h = (const float*)d_in[2];
    const float* W_ioux   = (const float*)d_in[3];
    const float* b_ioux   = (const float*)d_in[4];
    const float* W_iouh   = (const float*)d_in[5];
    const float* b_iouh   = (const float*)d_in[6];
    const float* W_ious   = (const float*)d_in[7];
    const float* b_ious   = (const float*)d_in[8];
    const float* W_fx     = (const float*)d_in[9];
    const float* b_fx     = (const float*)d_in[10];
    const float* W_fxs    = (const float*)d_in[11];
    const float* b_fxs    = (const float*)d_in[12];
    const float* W_fh     = (const float*)d_in[13];
    const float* b_fh     = (const float*)d_in[14];
    const float* W_fs     = (const float*)d_in[15];
    const float* b_fs     = (const float*)d_in[16];

    char* ws = (char*)d_ws;
    bf16_t* A       = (bf16_t*)(ws + OFF_A);
    bf16_t* W       = (bf16_t*)(ws + OFF_W);
    bf16_t* Wh      = (bf16_t*)(ws + OFF_WH);
    float*  bias    = (float*)(ws + OFF_BIAS);
    float*  c_state = (float*)(ws + OFF_C);
    bf16_t* h0      = (bf16_t*)(ws + OFF_H0);
    bf16_t* h1      = (bf16_t*)(ws + OFF_H1);
    float*  Pre     = (float*)(ws + OFF_PRE);
    float*  hs      = (float*)d_out;

    pack_A_kernel<<<(ROWS * KDIM) / 256, 256, 0, stream>>>(inputs, sememe_h, A);
    pack_W_kernel<<<(NCAT * KDIM) / 256, 256, 0, stream>>>(W_ioux, W_ious, W_fx, W_fxs, W_fs, W);
    pack_Wh_kernel<<<(2048 * MDIM) / 256, 256, 0, stream>>>(W_iouh, W_fh, Wh);
    pack_misc_kernel<<<(BATCH * MDIM) / 256, 256, 0, stream>>>(
        b_ioux, b_ious, b_iouh, b_fx, b_fh, b_fxs, b_fs, bias, c_state, h0, h1);

    gemm_pre_kernel<<<dim3(NCAT / 256, ROWS / 128), 256, 0, stream>>>(A, W, bias, sememe_c, Pre);

    for (int t = 0; t < T_STEPS; ++t) {
        bf16_t* hin  = (t & 1) ? h1 : h0;
        bf16_t* hout = (t & 1) ? h0 : h1;
        lstm_step_kernel<<<8, 256, 0, stream>>>(t, Pre, Wh, hin, hout, c_state, hs);
    }
}
